// TransMILExpert_12910671692267
// MI455X (gfx1250) — matrix operand order repaired and checked by reading, not compiled, not run on hardware
//
#include <hip/hip_runtime.h>
#include <hip/hip_bf16.h>

// ---------------------------------------------------------------------------
// TransMIL forward for MI455X (gfx1250, wave32, WMMA bf16 16x16x32)
// ---------------------------------------------------------------------------

typedef __bf16 bf16_t;
typedef __attribute__((ext_vector_type(16))) __bf16 v16bf;
typedef __attribute__((ext_vector_type(8)))  float  v8f;

#define NT   40001          // tokens incl. cls
#define N0   40000
#define NP   40192          // padded for attention (256*157)
#define PADR 191
#define DM   512
#define NH   4
#define DH   128
#define ML   256
#define LSEG 157

static __device__ __forceinline__ int a_kidx(int e, int half) {
    // 16-bit A-matrix 16x32 layout (contiguous, matches B): elem e -> K = half*16+e
    return half * 16 + e;
}

// ------------------------------ WMMA GEMM ----------------------------------
enum { EPI_F32 = 0, EPI_RELU_BIAS = 1, EPI_QKV = 2, EPI_RESID = 3 };

template <bool TRANSB, int EPI>
__global__ __launch_bounds__(128)
void wgemm(const bf16_t* __restrict__ A, int lda, long sA,
           const bf16_t* __restrict__ B, int ldb, long sB,
           float* __restrict__ C, int ldc, long sC,
           bf16_t* __restrict__ Cbf,
           const float* __restrict__ bias,
           float* __restrict__ resid, int rowoff,
           int M, int N, int K, float qscale)
{
    const int lane = threadIdx.x & 31, wv = threadIdx.x >> 5;
    const int half = lane >> 4, lm = lane & 15;
    const int m0 = (blockIdx.y * 4 + wv) * 16;
    const int n0 = blockIdx.x * 64;
    if (m0 >= M) return;
    const long bz = blockIdx.z;
    A += bz * sA; B += bz * sB; C += bz * sC;

    v8f acc[4] = {};
    for (int kk = 0; kk < K; kk += 32) {
        const bf16_t* Ap = A + (long)(m0 + lm) * lda + kk;
        __builtin_prefetch(Ap + 32, 0, 1);
        v16bf af;
#pragma unroll
        for (int e = 0; e < 16; ++e) af[e] = Ap[a_kidx(e, half)];
#pragma unroll
        for (int j = 0; j < 4; ++j) {
            const int n = n0 + j * 16 + lm;
            v16bf bfr;
            if (TRANSB) {
                const bf16_t* Bp = B + (long)n * ldb + kk + half * 16;
#pragma unroll
                for (int e = 0; e < 16; ++e) bfr[e] = Bp[e];
            } else {
                const bf16_t* Bp = B + (long)(kk + half * 16) * ldb + n;
#pragma unroll
                for (int e = 0; e < 16; ++e) bfr[e] = Bp[(long)e * ldb];
            }
            acc[j] = __builtin_amdgcn_wmma_f32_16x16x32_bf16(
                false, af, false, bfr, (short)0, acc[j], false, false);
        }
    }
#pragma unroll
    for (int j = 0; j < 4; ++j) {
#pragma unroll
        for (int r = 0; r < 8; ++r) {
            const int m = m0 + r + 8 * half;
            const int n = n0 + j * 16 + lm;
            const float a = acc[j][r];
            if (EPI == EPI_RELU_BIAS) {
                float o = a + bias[n];
                C[(long)m * ldc + n] = o > 0.f ? o : 0.f;
            } else if (EPI == EPI_QKV) {
                float o = a * (n < 512 ? qscale : 1.f);
                Cbf[(long)m * ldc + n] = (bf16_t)o;
            } else if (EPI == EPI_RESID) {
                const int row = m - rowoff;
                if (row >= 0) resid[(long)row * ldc + n] += a + bias[n];
            } else {
                C[(long)m * ldc + n] = a;
            }
        }
    }
}

// ------------------------- small f32 GEMM (pinv) ---------------------------
// out = c0*(A@B) + c1*Cin + c2*I       (batched over grid.z)
__global__ __launch_bounds__(256)
void sgemm_f32(const float* __restrict__ A, const float* __restrict__ B,
               float* __restrict__ C, const float* __restrict__ Cin,
               float c0, float c1, float c2,
               int M, int N, int K, long sA, long sB, long sC)
{
    __shared__ float As[16][17], Bs[16][17];
    const int tx = threadIdx.x, ty = threadIdx.y;
    const int m = blockIdx.y * 16 + ty, n = blockIdx.x * 16 + tx;
    const long bz = blockIdx.z;
    A += bz * sA; B += bz * sB; C += bz * sC;
    float acc = 0.f;
    for (int kk = 0; kk < K; kk += 16) {
        As[ty][tx] = A[(long)m * K + kk + tx];
        Bs[ty][tx] = B[(long)(kk + ty) * N + n];
        __syncthreads();
#pragma unroll
        for (int k = 0; k < 16; ++k) acc = fmaf(As[ty][k], Bs[k][tx], acc);
        __syncthreads();
    }
    float o = c0 * acc;
    if (Cin) o += c1 * Cin[bz * sC + (long)m * N + n];
    if (m == n) o += c2;
    C[(long)m * N + n] = o;
}

// ------------------------------ LayerNorm ----------------------------------
__global__ __launch_bounds__(256)
void ln_pad_bf16(const float* __restrict__ x, const float* __restrict__ g,
                 const float* __restrict__ b, bf16_t* __restrict__ xp)
{
    const int row = blockIdx.x;   // 0..NP-1
    const int t = threadIdx.x;
    if (row < PADR) {
        xp[(long)row * DM + t] = (bf16_t)0.f;
        xp[(long)row * DM + t + 256] = (bf16_t)0.f;
        return;
    }
    const float* src = x + (long)(row - PADR) * DM;
    float v0 = src[t], v1 = src[t + 256];
    __shared__ float red[256];
    red[t] = v0 + v1; __syncthreads();
    for (int s = 128; s > 0; s >>= 1) { if (t < s) red[t] += red[t + s]; __syncthreads(); }
    const float mu = red[0] * (1.f / DM);
    __syncthreads();
    const float d0 = v0 - mu, d1 = v1 - mu;
    red[t] = d0 * d0 + d1 * d1; __syncthreads();
    for (int s = 128; s > 0; s >>= 1) { if (t < s) red[t] += red[t + s]; __syncthreads(); }
    const float rsd = rsqrtf(red[0] * (1.f / DM) + 1e-5f);
    xp[(long)row * DM + t]       = (bf16_t)(d0 * rsd * g[t] + b[t]);
    xp[(long)row * DM + t + 256] = (bf16_t)(d1 * rsd * g[t + 256] + b[t + 256]);
}

// ------------------------------ landmarks ----------------------------------
__global__ __launch_bounds__(512)
void landmark_means(const bf16_t* __restrict__ qkv,
                    bf16_t* __restrict__ ql, bf16_t* __restrict__ kl)
{
    const int m = blockIdx.x, c = threadIdx.x;
    const bf16_t* base = qkv + (long)m * LSEG * 1536;
    float sq = 0.f, sk = 0.f;
    for (int j = 0; j < LSEG; ++j) {
        sq += (float)base[(long)j * 1536 + c];
        sk += (float)base[(long)j * 1536 + 512 + c];
    }
    ql[(long)m * DM + c] = (bf16_t)(sq * (1.f / LSEG));
    kl[(long)m * DM + c] = (bf16_t)(sk * (1.f / LSEG));
}

// ------------------------------ row softmax --------------------------------
__global__ __launch_bounds__(256)
void row_softmax(const float* __restrict__ S, float* __restrict__ Pf,
                 bf16_t* __restrict__ Pb, int cols)
{
    const long row = blockIdx.x;
    const float* s = S + row * (long)cols;
    __shared__ float red[256];
    const int t = threadIdx.x;
    float mx = -3.4e38f;
    for (int i = t; i < cols; i += 256) mx = fmaxf(mx, s[i]);
    red[t] = mx; __syncthreads();
    for (int st = 128; st > 0; st >>= 1) { if (t < st) red[t] = fmaxf(red[t], red[t + st]); __syncthreads(); }
    mx = red[0]; __syncthreads();
    float sum = 0.f;
    for (int i = t; i < cols; i += 256) sum += __expf(s[i] - mx);
    red[t] = sum; __syncthreads();
    for (int st = 128; st > 0; st >>= 1) { if (t < st) red[t] += red[t + st]; __syncthreads(); }
    const float inv = 1.f / red[0];
    for (int i = t; i < cols; i += 256) {
        const float p = __expf(s[i] - mx) * inv;
        if (Pf) Pf[row * (long)cols + i] = p;
        if (Pb) Pb[row * (long)cols + i] = (bf16_t)p;
    }
}

// ------------------------------ pinv helpers -------------------------------
__global__ __launch_bounds__(256)
void a2_abs_sums(const float* __restrict__ a2, float* __restrict__ rs, float* __restrict__ cs)
{
    const int idx = blockIdx.x, h = idx >> 8, i = idx & 255, t = threadIdx.x;
    const float* base = a2 + (long)h * 65536;
    __shared__ float r1[256], r2[256];
    r1[t] = fabsf(base[i * 256 + t]);
    r2[t] = fabsf(base[t * 256 + i]);
    __syncthreads();
    for (int s = 128; s > 0; s >>= 1) { if (t < s) { r1[t] += r1[t + s]; r2[t] += r2[t + s]; } __syncthreads(); }
    if (!t) { rs[idx] = r1[0]; cs[idx] = r2[0]; }
}

__global__ __launch_bounds__(1024)
void pinv_scale(const float* __restrict__ rs, const float* __restrict__ cs, float* __restrict__ scal)
{
    __shared__ float m1[1024], m2[1024];
    const int t = threadIdx.x;
    m1[t] = rs[t]; m2[t] = cs[t]; __syncthreads();
    for (int s = 512; s > 0; s >>= 1) { if (t < s) { m1[t] = fmaxf(m1[t], m1[t + s]); m2[t] = fmaxf(m2[t], m2[t + s]); } __syncthreads(); }
    if (!t) scal[0] = 1.f / (m1[0] * m2[0]);
}

__global__ __launch_bounds__(256)
void pinv_init(const float* __restrict__ a2, const float* __restrict__ scal, float* __restrict__ z)
{
    const int h = blockIdx.x, i = blockIdx.y, j = threadIdx.x;
    z[(long)h * 65536 + i * 256 + j] = a2[(long)h * 65536 + j * 256 + i] * scal[0];
}

// ------------------- fused a1: softmax(q@kl^T) @ Z2 ------------------------
__global__ __launch_bounds__(64)
void a1_fused(const bf16_t* __restrict__ qkv, const bf16_t* __restrict__ kl,
              const bf16_t* __restrict__ z2, float* __restrict__ attn)
{
    const int wv = threadIdx.x >> 5, lane = threadIdx.x & 31;
    const int half = lane >> 4, lm = lane & 15;
    const int head = blockIdx.y;
    const int m0 = blockIdx.x * 32 + wv * 16;
    __shared__ float  sS[2][16 * 256];
    __shared__ bf16_t sP[2][16 * 256];
    float*  S = sS[wv];
    bf16_t* P = sP[wv];

    // phase 1: scores 16 x 256 (K = 128)
    v8f acc[16] = {};
    for (int kk = 0; kk < 128; kk += 32) {
        const bf16_t* Ap = qkv + (long)(m0 + lm) * 1536 + head * DH + kk;
        v16bf af;
#pragma unroll
        for (int e = 0; e < 16; ++e) af[e] = Ap[a_kidx(e, half)];
#pragma unroll
        for (int j = 0; j < 16; ++j) {
            const bf16_t* Bp = kl + (long)(j * 16 + lm) * DM + head * DH + kk + half * 16;
            v16bf bfr;
#pragma unroll
            for (int e = 0; e < 16; ++e) bfr[e] = Bp[e];
            acc[j] = __builtin_amdgcn_wmma_f32_16x16x32_bf16(
                false, af, false, bfr, (short)0, acc[j], false, false);
        }
    }
#pragma unroll
    for (int j = 0; j < 16; ++j)
#pragma unroll
        for (int r = 0; r < 8; ++r)
            S[(r + 8 * half) * 256 + j * 16 + lm] = acc[j][r];
    __syncthreads();

    // softmax: 2 lanes per row (row = lm, half picks 128-col segment)
    const float* srow = S + lm * 256 + half * 128;
    float mx = -3.4e38f;
    for (int i = 0; i < 128; ++i) mx = fmaxf(mx, srow[i]);
    mx = fmaxf(mx, __shfl_xor(mx, 16));
    float sum = 0.f;
    for (int i = 0; i < 128; ++i) sum += __expf(srow[i] - mx);
    sum += __shfl_xor(sum, 16);
    const float inv = 1.f / sum;
    bf16_t* prow = P + lm * 256 + half * 128;
    for (int i = 0; i < 128; ++i) prow[i] = (bf16_t)(__expf(srow[i] - mx) * inv);
    __syncthreads();

    // phase 2: 16 x 128 = P(16x256) @ Z2(256x128)
    v8f o[8] = {};
    for (int kk = 0; kk < 256; kk += 32) {
        v16bf af;
#pragma unroll
        for (int e = 0; e < 16; ++e) af[e] = P[lm * 256 + kk + a_kidx(e, half)];
#pragma unroll
        for (int j = 0; j < 8; ++j) {
            const bf16_t* Bp = z2 + (long)head * 32768 + (long)(kk + half * 16) * DH + j * 16 + lm;
            v16bf bfr;
#pragma unroll
            for (int e = 0; e < 16; ++e) bfr[e] = Bp[(long)e * DH];
            o[j] = __builtin_amdgcn_wmma_f32_16x16x32_bf16(
                false, af, false, bfr, (short)0, o[j], false, false);
        }
    }
#pragma unroll
    for (int j = 0; j < 8; ++j)
#pragma unroll
        for (int r = 0; r < 8; ++r)
            attn[(long)(m0 + r + 8 * half) * DM + head * DH + j * 16 + lm] = o[j][r];
}

// ----------------------- depthwise residual conv (k=33) --------------------
__global__ __launch_bounds__(128)
void res_conv_add(const float* __restrict__ attn, const bf16_t* __restrict__ qkv,
                  const float* __restrict__ rw, bf16_t* __restrict__ attnbf)
{
    const int n = blockIdx.x, head = blockIdx.y, c = threadIdx.x;
    const int col = head * DH + c;
    float s = attn[(long)n * DM + col];
#pragma unroll
    for (int t = 0; t < 33; ++t) {
        const int src = n + t - 16;
        if (src >= 0 && src < NP)
            s = fmaf(rw[head * 33 + t], (float)qkv[(long)src * 1536 + 1024 + col], s);
    }
    attnbf[(long)n * DM + col] = (bf16_t)s;
}

// ------------------------------- PPEG --------------------------------------
__global__ __launch_bounds__(128)
void ppeg_kernel(const float* __restrict__ xin, float* __restrict__ xout,
                 const float* __restrict__ w7, const float* __restrict__ b7,
                 const float* __restrict__ w5, const float* __restrict__ b5,
                 const float* __restrict__ w3, const float* __restrict__ b3)
{
    const int n = blockIdx.x;                       // 0..39999
    const int c = blockIdx.y * 128 + threadIdx.x;   // channel
    const int hh = n / 200, ww = n % 200;
    float acc = xin[(long)(1 + n) * DM + c] + b7[c] + b5[c] + b3[c];
    for (int dy = 0; dy < 7; ++dy) {
        const int yy = hh + dy - 3; if (yy < 0 || yy >= 200) continue;
        for (int dx = 0; dx < 7; ++dx) {
            const int xx = ww + dx - 3; if (xx < 0 || xx >= 200) continue;
            acc = fmaf(xin[(long)(1 + yy * 200 + xx) * DM + c], w7[c * 49 + dy * 7 + dx], acc);
        }
    }
    for (int dy = 0; dy < 5; ++dy) {
        const int yy = hh + dy - 2; if (yy < 0 || yy >= 200) continue;
        for (int dx = 0; dx < 5; ++dx) {
            const int xx = ww + dx - 2; if (xx < 0 || xx >= 200) continue;
            acc = fmaf(xin[(long)(1 + yy * 200 + xx) * DM + c], w5[c * 25 + dy * 5 + dx], acc);
        }
    }
    for (int dy = 0; dy < 3; ++dy) {
        const int yy = hh + dy - 1; if (yy < 0 || yy >= 200) continue;
        for (int dx = 0; dx < 3; ++dx) {
            const int xx = ww + dx - 1; if (xx < 0 || xx >= 200) continue;
            acc = fmaf(xin[(long)(1 + yy * 200 + xx) * DM + c], w3[c * 9 + dy * 3 + dx], acc);
        }
    }
    xout[(long)(1 + n) * DM + c] = acc;
}

// ----------------------------- small utils ---------------------------------
__global__ void f32_to_bf16(const float* __restrict__ in, bf16_t* __restrict__ out, long n)
{
    long i = (long)blockIdx.x * 256 + threadIdx.x;
    const long stride = (long)gridDim.x * 256;
    for (; i < n; i += stride) out[i] = (bf16_t)in[i];
}

__global__ void copy_f32(const float* __restrict__ s, float* __restrict__ d, int n)
{
    const int i = blockIdx.x * 256 + threadIdx.x;
    if (i < n) d[i] = s[i];
}

// ------------------------------ final head ---------------------------------
__global__ __launch_bounds__(256)
void cls_head(const float* __restrict__ x, const float* __restrict__ g,
              const float* __restrict__ b, const float* __restrict__ wcls,
              const float* __restrict__ bcls, float* __restrict__ out)
{
    const int t = threadIdx.x;
    __shared__ float red[256];
    const float v0 = x[t], v1 = x[t + 256];
    red[t] = v0 + v1; __syncthreads();
    for (int s = 128; s > 0; s >>= 1) { if (t < s) red[t] += red[t + s]; __syncthreads(); }
    const float mu = red[0] * (1.f / DM);
    __syncthreads();
    const float d0 = v0 - mu, d1 = v1 - mu;
    red[t] = d0 * d0 + d1 * d1; __syncthreads();
    for (int s = 128; s > 0; s >>= 1) { if (t < s) red[t] += red[t + s]; __syncthreads(); }
    const float rsd = rsqrtf(red[0] * (1.f / DM) + 1e-5f);
    __syncthreads();
    const float xn0 = d0 * rsd * g[t] + b[t];
    const float xn1 = d1 * rsd * g[t + 256] + b[t + 256];
    red[t] = xn0 * wcls[t * 2] + xn1 * wcls[(t + 256) * 2]; __syncthreads();
    for (int s = 128; s > 0; s >>= 1) { if (t < s) red[t] += red[t + s]; __syncthreads(); }
    if (!t) out[0] = red[0] + bcls[0];
    __syncthreads();
    red[t] = xn0 * wcls[t * 2 + 1] + xn1 * wcls[(t + 256) * 2 + 1]; __syncthreads();
    for (int s = 128; s > 0; s >>= 1) { if (t < s) red[t] += red[t + s]; __syncthreads(); }
    if (!t) out[1] = red[0] + bcls[1];
}

// ===========================================================================
extern "C" void kernel_launch(void* const* d_in, const int* in_sizes, int n_in,
                              void* d_out, int out_size, void* d_ws, size_t ws_size,
                              hipStream_t stream)
{
    const float* h        = (const float*)d_in[0];
    const float* w_embed  = (const float*)d_in[1];
    const float* b_embed  = (const float*)d_in[2];
    const float* cls_tok  = (const float*)d_in[3];
    const float* norm_g   = (const float*)d_in[4];
    const float* norm_b   = (const float*)d_in[5];
    const float* wqkv     = (const float*)d_in[6];
    const float* wout     = (const float*)d_in[7];
    const float* bout     = (const float*)d_in[8];
    const float* res_w    = (const float*)d_in[9];
    const float* w7       = (const float*)d_in[10];
    const float* b7       = (const float*)d_in[11];
    const float* w5       = (const float*)d_in[12];
    const float* b5       = (const float*)d_in[13];
    const float* w3       = (const float*)d_in[14];
    const float* b3       = (const float*)d_in[15];
    const float* fnorm_g  = (const float*)d_in[16];
    const float* fnorm_b  = (const float*)d_in[17];
    const float* wcls     = (const float*)d_in[18];
    const float* bcls     = (const float*)d_in[19];
    float* outp = (float*)d_out;

    // ---------------- workspace layout (deterministic cursor) --------------
    char* w = (char*)d_ws;
    auto alloc = [&](size_t bytes) -> char* {
        char* p = w; w += (bytes + 255) & ~(size_t)255; return p;
    };
    float* xA   = (float*)alloc((size_t)NT * DM * 4);
    float* xB   = (float*)alloc((size_t)NT * DM * 4);
    float* S2   = (float*)alloc((size_t)NH * 65536 * 4);
    float* a2f  = (float*)alloc((size_t)NH * 65536 * 4);
    float* zA   = (float*)alloc((size_t)NH * 65536 * 4);
    float* zB   = (float*)alloc((size_t)NH * 65536 * 4);
    float* xz   = (float*)alloc((size_t)NH * 65536 * 4);
    float* p1   = (float*)alloc((size_t)NH * 65536 * 4);
    float* p2   = (float*)alloc((size_t)NH * 65536 * 4);
    float* rs   = (float*)alloc(1024 * 4);
    float* cs   = (float*)alloc(1024 * 4);
    float* scal = (float*)alloc(256);
    float* a3v  = (float*)alloc((size_t)NH * 32768 * 4);
    float* Z2   = (float*)alloc((size_t)NH * 32768 * 4);
    float* S3   = (float*)alloc((size_t)NH * ML * NP * 4);   // also aliased as attn_out
    bf16_t* webf   = (bf16_t*)alloc((size_t)1024 * 512 * 2);
    bf16_t* wqkvbf = (bf16_t*)alloc((size_t)2 * 512 * 1536 * 2);
    bf16_t* woutbf = (bf16_t*)alloc((size_t)2 * 512 * 512 * 2);
    bf16_t* xpbf   = (bf16_t*)alloc((size_t)NP * DM * 2);
    bf16_t* qkvbf  = (bf16_t*)alloc((size_t)NP * 1536 * 2);
    bf16_t* qlbf   = (bf16_t*)alloc((size_t)ML * DM * 2);
    bf16_t* klbf   = (bf16_t*)alloc((size_t)ML * DM * 2);
    bf16_t* Z2bf   = (bf16_t*)alloc((size_t)NH * 32768 * 2);
    bf16_t* attnbf = (bf16_t*)alloc((size_t)NP * DM * 2);
    bf16_t* a3bf   = (bf16_t*)alloc((size_t)NH * ML * NP * 2); // h_bf aliases this
    bf16_t* h_bf   = a3bf;           // embed input (dead before a3bf is written)
    float*  attn_out = S3;           // S3 is dead once a3bf is built

    const float qscale = 0.08838834764831845f; // 128^-0.5

    // ---------------- weight / input conversion ----------------------------
    f32_to_bf16<<<2048, 256, 0, stream>>>(h, h_bf, (long)N0 * 1024);
    f32_to_bf16<<<512, 256, 0, stream>>>(w_embed, webf, (long)1024 * 512);
    f32_to_bf16<<<512, 256, 0, stream>>>(wqkv, wqkvbf, (long)2 * 512 * 1536);
    f32_to_bf16<<<512, 256, 0, stream>>>(wout, woutbf, (long)2 * 512 * 512);
    copy_f32<<<2, 256, 0, stream>>>(cls_tok, xA, 512);   // cls -> row 0

    // ---------------- embed: relu(h @ w_embed + b_embed) -> xA rows 1.. ----
    wgemm<false, EPI_RELU_BIAS><<<dim3(8, 625, 1), 128, 0, stream>>>(
        h_bf, 1024, 0, webf, 512, 0, xA + DM, DM, 0,
        nullptr, b_embed, nullptr, 0, N0, DM, 1024, 1.f);

    float* cur = xA;
    for (int l = 0; l < 2; ++l) {
        // LN -> padded bf16
        ln_pad_bf16<<<NP, 256, 0, stream>>>(cur, norm_g + l * DM, norm_b + l * DM, xpbf);
        // QKV (q scaled)
        wgemm<false, EPI_QKV><<<dim3(24, NP / 64, 1), 128, 0, stream>>>(
            xpbf, DM, 0, wqkvbf + (size_t)l * 512 * 1536, 1536, 0,
            nullptr, 1536, 0, qkvbf, nullptr, nullptr, 0, NP, 1536, DM, qscale);
        // landmarks
        landmark_means<<<ML, 512, 0, stream>>>(qkvbf, qlbf, klbf);
        // a2 = softmax(ql @ kl^T)
        wgemm<true, EPI_F32><<<dim3(4, 4, NH), 128, 0, stream>>>(
            qlbf, DM, DH, klbf, DM, DH, S2, 256, 65536,
            nullptr, nullptr, nullptr, 0, ML, ML, DH, 1.f);
        row_softmax<<<NH * ML, 256, 0, stream>>>(S2, a2f, nullptr, 256);
        // pinv init
        a2_abs_sums<<<NH * ML, 256, 0, stream>>>(a2f, rs, cs);
        pinv_scale<<<1, 1024, 0, stream>>>(rs, cs, scal);
        pinv_init<<<dim3(NH, ML), 256, 0, stream>>>(a2f, scal, zA);
        // Newton-Schulz: z = 0.25 z (13I - xz(15I - xz(7I - xz)))
        float* zc = zA; float* zn = zB;
        for (int it = 0; it < 6; ++it) {
            sgemm_f32<<<dim3(16, 16, NH), dim3(16, 16), 0, stream>>>(
                a2f, zc, xz, nullptr, 1.f, 0.f, 0.f, 256, 256, 256, 65536, 65536, 65536);
            sgemm_f32<<<dim3(16, 16, NH), dim3(16, 16), 0, stream>>>(
                xz, xz, p1, xz, 1.f, -7.f, 15.f, 256, 256, 256, 65536, 65536, 65536);
            sgemm_f32<<<dim3(16, 16, NH), dim3(16, 16), 0, stream>>>(
                xz, p1, p2, nullptr, -1.f, 0.f, 13.f, 256, 256, 256, 65536, 65536, 65536);
            sgemm_f32<<<dim3(16, 16, NH), dim3(16, 16), 0, stream>>>(
                zc, p2, zn, nullptr, 0.25f, 0.f, 0.f, 256, 256, 256, 65536, 65536, 65536);
            float* t = zc; zc = zn; zn = t;
        }
        // S3 = ql @ k^T ; a3 = softmax rows
        wgemm<true, EPI_F32><<<dim3(NP / 64, 4, NH), 128, 0, stream>>>(
            qlbf, DM, DH, qkvbf + 512, 1536, DH, S3, NP, (long)ML * NP,
            nullptr, nullptr, nullptr, 0, ML, NP, DH, 1.f);
        row_softmax<<<NH * ML, 256, 0, stream>>>(S3, nullptr, a3bf, NP);
        // a3v = a3 @ v
        wgemm<false, EPI_F32><<<dim3(2, 4, NH), 128, 0, stream>>>(
            a3bf, NP, (long)ML * NP, qkvbf + 1024, 1536, DH, a3v, DH, 32768,
            nullptr, nullptr, nullptr, 0, ML, DH, NP, 1.f);
        // Z2 = z @ a3v
        sgemm_f32<<<dim3(8, 16, NH), dim3(16, 16), 0, stream>>>(
            zc, a3v, Z2, nullptr, 1.f, 0.f, 0.f, 256, 128, 256, 65536, 32768, 32768);
        f32_to_bf16<<<512, 256, 0, stream>>>(Z2, Z2bf, (long)NH * 32768);
        // attn = softmax(q @ kl^T) @ Z2
        a1_fused<<<dim3(NP / 32, NH), 64, 0, stream>>>(qkvbf, klbf, Z2bf, attn_out);
        // + depthwise residual conv on v
        res_conv_add<<<dim3(NP, NH), 128, 0, stream>>>(
            attn_out, qkvbf, res_w + l * NH * 33, attnbf);
        // out @ wout + bout, residual add into cur (last NT rows)
        wgemm<false, EPI_RESID><<<dim3(8, NP / 64, 1), 128, 0, stream>>>(
            attnbf, DM, 0, woutbf + (size_t)l * 512 * 512, DM, 0,
            nullptr, DM, 0, nullptr, bout + l * DM, cur, PADR, NP, DM, DM, 1.f);
        // PPEG after layer 0
        if (l == 0) {
            copy_f32<<<2, 256, 0, stream>>>(cur, xB, 512);  // cls row
            ppeg_kernel<<<dim3(N0, 4), 128, 0, stream>>>(cur, xB, w7, b7, w5, b5, w3, b3);
            cur = xB;
        }
    }
    // final LN(row 0) @ wcls + bcls
    cls_head<<<1, 256, 0, stream>>>(cur, fnorm_g, fnorm_b, wcls, bcls, outp);
}
